// WTConv2d_CPP_46514495816359
// MI455X (gfx1250) — matrix operand order repaired and checked by reading, not compiled, not run on hardware
//
#include <hip/hip_runtime.h>
#include <hip/hip_bf16.h>

// WTConv2d for gfx1250 (MI455X): Haar DWT -> implicit-GEMM 5x5 conv via
// v_wmma_f32_16x16x32_f16 -> Haar IDWT reconstruction.
//
// Conv as GEMM per level: M = B*H2*W2 pixels, N = 256 out-channels,
// K = 25 taps * 256 in-channels.
//   A = NHWC f16 patches, double-buffered through LDS (half-tap slabs),
//       stored in plain k order so each lane's WMMA fragment is contiguous,
//   B = pre-transposed f16 weights wt[o][tap*256+ci] straight from L2,
//   2x2 register blocking per wave32 (4 f32 accumulators), f32 accumulate.

typedef __attribute__((ext_vector_type(16))) _Float16 v16h;
typedef __attribute__((ext_vector_type(8)))  _Float16 h8;
typedef __attribute__((ext_vector_type(8)))  float    v8f;

#define WT_C    64
#define WT_C4   256
#define WT_K    6400   // 25 * 256

// ---------------------------------------------------------------------------
// Weight prep: w[o][i][kh][kw] f32 -> wt[o][ (kh*5+kw)*256 + i ] f16
// ---------------------------------------------------------------------------
__global__ void wt_prep_w(const float* __restrict__ w, _Float16* __restrict__ wt) {
  int idx = blockIdx.x * blockDim.x + threadIdx.x;
  if (idx >= WT_C4 * WT_K) return;
  int o = idx / WT_K;
  int k = idx - o * WT_K;
  int t = k >> 8;        // tap 0..24
  int i = k & 255;       // in-channel
  wt[idx] = (_Float16)w[((size_t)o * WT_C4 + i) * 25 + t];
}

// ---------------------------------------------------------------------------
// Haar DWT + pack: cur f32 [B,64,2H2,2W2] ->
//   sp  f16 NHWC [B,H2,W2,256] with channel ci = c*4 + {ll,lh,hl,hh}
//   llout f32 NCHW [B,64,H2,W2] (next level input; may be null)
// ---------------------------------------------------------------------------
__global__ void wt_dwt_pack(const float* __restrict__ cur, _Float16* __restrict__ sp,
                            float* __restrict__ llout, int B, int H2, int W2) {
  int idx = blockIdx.x * blockDim.x + threadIdx.x;
  int total = B * H2 * W2 * WT_C;
  if (idx >= total) return;
  int c = idx & 63;
  int p = idx >> 6;                 // pixel index (b*H2 + y)*W2 + x
  int x = p % W2;
  int r = p / W2;
  int yrow = r % H2;
  int b = r / H2;

  const int W = W2 * 2;
  const float* src = cur + ((size_t)(b * WT_C + c) * (H2 * 2) + 2 * yrow) * W + 2 * x;
  float a  = src[0];
  float b2 = src[1];
  float c2 = src[W];
  float d2 = src[W + 1];

  float ll = 0.5f * (a + b2 + c2 + d2);
  float lh = 0.5f * (a + b2 - c2 - d2);
  float hl = 0.5f * (a - b2 + c2 - d2);
  float hh = 0.5f * (a - b2 - c2 + d2);

  _Float16* o = sp + (size_t)p * WT_C4 + c * 4;
  o[0] = (_Float16)ll;
  o[1] = (_Float16)lh;
  o[2] = (_Float16)hl;
  o[3] = (_Float16)hh;
  if (llout) llout[((size_t)(b * WT_C + c) * H2 + yrow) * W2 + x] = ll;
}

// ---------------------------------------------------------------------------
// Implicit-GEMM conv, WMMA f16->f32, 2x2 register blocking per wave.
//   sp : f16 NHWC [B,H2,W2,256]   (A operand, zero-padded at borders)
//   wt : f16 [256][6400]          (B operand, column o = wt row, contiguous k)
//   y  : f32 NHWC [B,H2,W2,256]
// H2 == W2 == 1<<L (128/64/32). Block = 256 thr (8 waves) -> 64 px x 128 ch.
// Wave w: m-pair = w&1 (rows 32*(w&1)..+31), n-pair = w>>1 (cols 32*(w>>1)..+31).
// Pipeline: 50 stages (25 taps x 2 half-taps of 128 ci), LDS double buffered.
// LDS layout per 32-K block: stored in plain k order [0..31] so each
// lane half reads one contiguous 32B run for its A fragment.
// ---------------------------------------------------------------------------
__global__ __launch_bounds__(256)
void wt_conv_wmma(const _Float16* __restrict__ sp, const _Float16* __restrict__ wt,
                  float* __restrict__ y, int L) {
  __shared__ __align__(16) _Float16 As[2][64][136];  // 2 x 17KB, padded stride

  const int tid  = threadIdx.x;
  const int lane = tid & 31;
  const int wv   = tid >> 5;
  const int lm   = lane & 15;     // A row / B column / D column index
  const int half = lane >> 4;     // K-half selector per WMMA lane layout
  const int mp   = wv & 1;        // m-pair (rows 32*mp .. 32*mp+31)
  const int np   = wv >> 1;       // n-pair 0..3 (cols 32*np .. 32*np+31)

  const int W2   = 1 << L;
  const int mskW = W2 - 1;
  const int HW   = 1 << (2 * L);

  const int pixbase = blockIdx.x << 6;                 // 64 pixels per block
  const int nbase   = (blockIdx.y << 7) + (np << 5);   // two 16-wide N slices

  const _Float16* wrow0 = wt + (size_t)(nbase + lm) * WT_K + half * 16;
  const _Float16* wrow1 = wrow0 + (size_t)16 * WT_K;

  v8f acc00 = {}, acc01 = {}, acc10 = {}, acc11 = {};

  // ---- loop-invariant staging addresses (4 h8 chunks per thread) ----------
  const _Float16* gbase[4];   // global source base (no tap shift, slab cb=0)
  _Float16*       lbase0[4];  // LDS dest, buffer 0 (column c8, no swizzle)
  _Float16*       lbase1[4];  // LDS dest, buffer 1
  int xx4[4], yy4[4];
  #pragma unroll
  for (int it = 0; it < 4; ++it) {
    const int e    = tid + (it << 8);   // 0..1023
    const int px   = e >> 4;            // 0..63
    const int c8   = (e & 15) << 3;     // 0..120 step 8
    const int g    = (c8 >> 3) & 3;             // chunk within 32-K block
    const int gs   = g;                         // identity -> {0,1,2,3}
    const int c8sw = (c8 & ~31) | (gs << 3);    // LDS column, equal to c8
    const int p    = pixbase + px;
    const int x    = p & mskW;
    const int row  = p >> L;
    const int yy   = row & mskW;
    const int bb   = row >> L;
    gbase[it]  = sp + ((size_t)bb * HW + (size_t)yy * W2 + x) * WT_C4 + c8;
    lbase0[it] = &As[0][px][c8sw];
    lbase1[it] = &As[1][px][c8sw];
    xx4[it] = x;
    yy4[it] = yy;
  }

  // Stage s (0..49): tap t = s>>1, ci slab cb = (s&1)*128 -> As[buf][64][128].
  auto stage = [&](int s, int buf) {
    const int t    = s >> 1;
    const int cb   = (s & 1) << 7;
    const int dy   = t / 5 - 2;
    const int dx   = (t - 5 * (t / 5)) - 2;
    const int goff = (dy * W2 + dx) * WT_C4 + cb;   // wave-uniform scalar
    #pragma unroll
    for (int it = 0; it < 4; ++it) {
      const unsigned sx = (unsigned)(xx4[it] + dx);
      const unsigned sy = (unsigned)(yy4[it] + dy);
      h8 v = {};
      if (sx < (unsigned)W2 && sy < (unsigned)W2) {
        v = *(const h8*)(gbase[it] + goff);
      }
      *(h8*)((buf ? lbase1 : lbase0)[it]) = v;
    }
  };

  stage(0, 0);

  const int r0   = (mp << 5) + lm;   // LDS row of first m-subtile
  const int r1   = r0 + 16;          // second m-subtile
  const int coff = half << 4;        // column base for this lane half (k = 16*half)

  const _Float16* wk0 = wrow0;       // k offset of stage s is exactly s*128
  const _Float16* wk1 = wrow1;

  for (int s = 0; s < 50; ++s) {
    const int buf = s & 1;
    __syncthreads();                   // slab s staged by all waves
    if (s < 49) stage(s + 1, buf ^ 1); // issue next slab's loads -> overlap WMMAs

    __builtin_prefetch(wk0 + 128, 0, 0);   // next slab's weights

    #pragma unroll
    for (int kc = 0; kc < 128; kc += 32) {
      // A 16x32 f16 frags: plain-order LDS -> one contiguous 32B run per lane
      h8 a0lo = *(const h8*)&As[buf][r0][kc + coff];
      h8 a0hi = *(const h8*)&As[buf][r0][kc + coff + 8];
      h8 a1lo = *(const h8*)&As[buf][r1][kc + coff];
      h8 a1hi = *(const h8*)&As[buf][r1][kc + coff + 8];
      v16h a0 = __builtin_shufflevector(a0lo, a0hi, 0, 1, 2, 3, 4, 5, 6, 7,
                                                    8, 9, 10, 11, 12, 13, 14, 15);
      v16h a1 = __builtin_shufflevector(a1lo, a1hi, 0, 1, 2, 3, 4, 5, 6, 7,
                                                    8, 9, 10, 11, 12, 13, 14, 15);
      // B 32x16 f16 frags: lane holds column, contiguous 16 K values
      v16h b0 = *(const v16h*)(wk0 + kc);
      v16h b1 = *(const v16h*)(wk1 + kc);
      acc00 = __builtin_amdgcn_wmma_f32_16x16x32_f16(false, a0, false, b0,
                                                     (short)0, acc00, false, false);
      acc01 = __builtin_amdgcn_wmma_f32_16x16x32_f16(false, a0, false, b1,
                                                     (short)0, acc01, false, false);
      acc10 = __builtin_amdgcn_wmma_f32_16x16x32_f16(false, a1, false, b0,
                                                     (short)0, acc10, false, false);
      acc11 = __builtin_amdgcn_wmma_f32_16x16x32_f16(false, a1, false, b1,
                                                     (short)0, acc11, false, false);
    }
    wk0 += 128;
    wk1 += 128;
  }

  // D 16x16 f32: VGPR v -> row half*8+v, column lm.
  auto store_tile = [&](const v8f& a, int msub, int nsub) {
    float* yo = y + (size_t)(pixbase + ((mp * 2 + msub) << 4)) * WT_C4
                  + nbase + (nsub << 4) + lm;
    #pragma unroll
    for (int v = 0; v < 8; ++v) {
      yo[(size_t)(half * 8 + v) * WT_C4] = a[v];
    }
  };
  store_tile(acc00, 0, 0);
  store_tile(acc01, 0, 1);
  store_tile(acc10, 1, 0);
  store_tile(acc11, 1, 1);
}

// ---------------------------------------------------------------------------
// Haar IDWT + residual add (+ optional bias on final level):
//   yN   : f32 NHWC [B,H2,W2,256] conv output (o = c*4 + subband)
//   prev : f32 NCHW [B,64,H2,W2] next_ll from deeper level (null -> 0)
//   out  : f32 NCHW [B,64,2H2,2W2]
// ---------------------------------------------------------------------------
__global__ void wt_idwt(const float* __restrict__ yN, const float* __restrict__ prev,
                        const float* __restrict__ bias, float* __restrict__ out,
                        int B, int H2, int W2) {
  int idx = blockIdx.x * blockDim.x + threadIdx.x;
  int total = B * WT_C * H2 * W2;
  if (idx >= total) return;
  int x = idx % W2;
  int r = idx / W2;
  int yrow = r % H2;
  r /= H2;
  int c = r % WT_C;
  int b = r / WT_C;

  const float4 v = *(const float4*)(yN + ((size_t)(b * H2 + yrow) * W2 + x) * WT_C4 + c * 4);
  float ll = v.x + (prev ? prev[idx] : 0.0f);
  float lh = v.y, hl = v.z, hh = v.w;

  float a  = 0.5f * (ll + lh + hl + hh);
  float b2 = 0.5f * (ll + lh - hl - hh);
  float c2 = 0.5f * (ll - lh + hl - hh);
  float d2 = 0.5f * (ll - lh - hl + hh);
  if (bias) {
    float bv = bias[c];
    a += bv; b2 += bv; c2 += bv; d2 += bv;
  }

  const int W = W2 * 2;
  float* o = out + ((size_t)(b * WT_C + c) * (H2 * 2) + 2 * yrow) * W + 2 * x;
  o[0]     = a;
  o[1]     = b2;
  o[W]     = c2;
  o[W + 1] = d2;
}

// ---------------------------------------------------------------------------
extern "C" void kernel_launch(void* const* d_in, const int* in_sizes, int n_in,
                              void* d_out, int out_size, void* d_ws, size_t ws_size,
                              hipStream_t stream) {
  (void)in_sizes; (void)n_in; (void)out_size; (void)ws_size;

  const float* x    = (const float*)d_in[0];
  const float* w[3] = {(const float*)d_in[1], (const float*)d_in[2], (const float*)d_in[3]};
  const float* bias = (const float*)d_in[4];
  float* out = (float*)d_out;
  char* ws = (char*)d_ws;

  const int B = 8;

  // Workspace layout (256B aligned). Total ~316 MB with aliasing below.
  size_t off = 0;
  auto carve = [&](size_t bytes) {
    size_t o = off;
    off += (bytes + 255) & ~(size_t)255;
    return o;
  };
  const size_t o_s0  = carve((size_t)B * 128 * 128 * WT_C4 * 2);  // 67 MB f16 NHWC
  const size_t o_s1  = carve((size_t)B * 64 * 64 * WT_C4 * 2);    // 16.8 MB
  const size_t o_s2  = carve((size_t)B * 32 * 32 * WT_C4 * 2);    // 4.2 MB
  const size_t o_ll1 = carve((size_t)B * WT_C * 128 * 128 * 4);   // 33.5 MB
  const size_t o_ll2 = carve((size_t)B * WT_C * 64 * 64 * 4);     // 8.4 MB
  const size_t o_wt0 = carve((size_t)WT_C4 * WT_K * 2);           // 3.3 MB
  const size_t o_wt1 = carve((size_t)WT_C4 * WT_K * 2);
  const size_t o_wt2 = carve((size_t)WT_C4 * WT_K * 2);
  const size_t o_y0  = carve((size_t)B * 128 * 128 * WT_C4 * 4);  // 134 MB
  const size_t o_y1  = carve((size_t)B * 64 * 64 * WT_C4 * 4);    // 33.5 MB
  const size_t o_y2  = carve((size_t)B * 32 * 32 * WT_C4 * 4);    // 8.4 MB
  // Aliased scratch: s1/s0 regions are dead once conv1/conv0 have consumed them.
  const size_t o_r2  = o_s1;   // f32 [B,64,64,64]   (8.4 MB  <= 16.8 MB region)
  const size_t o_r1  = o_s0;   // f32 [B,64,128,128] (33.5 MB <= 67 MB region)

  _Float16* s0 = (_Float16*)(ws + o_s0);
  _Float16* s1 = (_Float16*)(ws + o_s1);
  _Float16* s2 = (_Float16*)(ws + o_s2);
  float* ll1 = (float*)(ws + o_ll1);
  float* ll2 = (float*)(ws + o_ll2);
  _Float16* wtp[3] = {(_Float16*)(ws + o_wt0), (_Float16*)(ws + o_wt1), (_Float16*)(ws + o_wt2)};
  float* y0 = (float*)(ws + o_y0);
  float* y1 = (float*)(ws + o_y1);
  float* y2 = (float*)(ws + o_y2);
  float* r2 = (float*)(ws + o_r2);
  float* r1 = (float*)(ws + o_r1);

  // Weight f32 -> f16 transpose (tiny, overlaps nothing it reads).
  for (int l = 0; l < 3; ++l) {
    wt_prep_w<<<(WT_C4 * WT_K + 255) / 256, 256, 0, stream>>>(w[l], wtp[l]);
  }

  // Forward: DWT+pack, then WMMA conv, three levels (L = log2(H2)).
  wt_dwt_pack<<<(B * 128 * 128 * WT_C) / 256, 256, 0, stream>>>(x,   s0, ll1, B, 128, 128);
  wt_conv_wmma<<<dim3((B * 128 * 128) / 64, 2), 256, 0, stream>>>(s0, wtp[0], y0, 7);

  wt_dwt_pack<<<(B * 64 * 64 * WT_C) / 256, 256, 0, stream>>>(ll1, s1, ll2, B, 64, 64);
  wt_conv_wmma<<<dim3((B * 64 * 64) / 64, 2), 256, 0, stream>>>(s1, wtp[1], y1, 6);

  wt_dwt_pack<<<(B * 32 * 32 * WT_C) / 256, 256, 0, stream>>>(ll2, s2, nullptr, B, 32, 32);
  wt_conv_wmma<<<dim3((B * 32 * 32) / 64, 2), 256, 0, stream>>>(s2, wtp[2], y2, 5);

  // Reconstruction: IDWT with residual add; bias folded into the last level.
  wt_idwt<<<(B * WT_C * 32 * 32) / 256, 256, 0, stream>>>(y2, nullptr, nullptr, r2, B, 32, 32);
  wt_idwt<<<(B * WT_C * 64 * 64) / 256, 256, 0, stream>>>(y1, r2, nullptr, r1, B, 64, 64);
  wt_idwt<<<(B * WT_C * 128 * 128) / 256, 256, 0, stream>>>(y0, r1, bias, out, B, 128, 128);
}
